// LSTM_1297080123470
// MI455X (gfx1250) — matrix operand order repaired and checked by reading, not compiled, not run on hardware
//
#include <hip/hip_runtime.h>
#include <cstdint>
#include <cstddef>

// ---------------------------------------------------------------------------
// 2-layer LSTM forward + 59-step autoregressive decode, CDNA5 (gfx1250).
//
//  - All matmuls: v_wmma_f32_16x16x32_bf16 (f32 accumulate).
//  - Weights pre-swizzled once into WMMA B-fragment order (contiguous 32B
//    per lane -> 2x global_load_b128 / 2x ds_load_b128).
//  - Input projections hoisted into two large GEMMs (M = B*T = 32768).
//  - Sequential recurrence: one fused kernel per step. 8-wave WG owns a
//    16-wide j-slice of H for ALL four gates; B fragments are double-buffered
//    through LDS (global_load_b128 -> ds_store_b128 staging overlapped with
//    WMMA consumption) so each weight byte is read from L2 exactly once per
//    step. Gate nonlinearities + c/h update fused in-register.
// ---------------------------------------------------------------------------

#define BB   128          // batch
#define TT   256          // prefix timesteps
#define DD   256          // input / output dim
#define HH   1024         // hidden
#define NG   4096         // 4*H (gate dim)
#define MTOT (BB * TT)    // 32768
#define DEC  59

typedef __bf16 bf16;
typedef __attribute__((ext_vector_type(16))) __bf16 v16bf;
typedef __attribute__((ext_vector_type(8)))  __bf16 bf16x8;
typedef __attribute__((ext_vector_type(8)))  float  v8f;

// ---- scalar helpers -------------------------------------------------------

__device__ __forceinline__ unsigned short f2bf_bits(float f) {
  unsigned u = __float_as_uint(f);
  unsigned r = 0x7FFFu + ((u >> 16) & 1u);   // round-to-nearest-even
  return (unsigned short)((u + r) >> 16);
}
__device__ __forceinline__ float bf2f(unsigned short b) {
  return __uint_as_float(((unsigned)b) << 16);
}
__device__ __forceinline__ float sigm(float x) { return 1.0f / (1.0f + __expf(-x)); }

// ---- WMMA fragment loaders ------------------------------------------------
// A (16x32 bf16, M x K): lanes 0-15 hold rows M=0..15 with K = k0+{0..15},
// lanes 16-31 same rows with K = k0+{16..31}.  Two 16B loads per lane.
__device__ __forceinline__ v16bf load_A_frag(const bf16* A, size_t lda,
                                             int m0, int k0, int lane) {
  const int row = m0 + (lane & 15);
  const int kb  = k0 + ((lane >> 4) << 4);
  const bf16* p = A + (size_t)row * lda + kb;
  union { v16bf v; bf16x8 h[2]; } u;
  u.h[0] = *(const bf16x8*)(p);
  u.h[1] = *(const bf16x8*)(p + 8);
  return u.v;
}
// B (32x16 bf16, K x N) pre-swizzled: block[ntile][kchunk] is 1KB, lane's
// 16 bf16 are contiguous 32B -> one v16bf (2x b128) load.
__device__ __forceinline__ v16bf load_B_frag(const bf16* WF, int ntile, int kc,
                                             int kchunks, int lane) {
  const bf16* p = WF + ((((size_t)ntile * kchunks) + kc) << 9) + (lane << 4);
  return *(const v16bf*)p;
}

#define WMMA_BF16(a, b, c) \
  __builtin_amdgcn_wmma_f32_16x16x32_bf16(false, (a), false, (b), (short)0, (c), false, false)

// ---- small utility kernels ------------------------------------------------

__global__ void zero_kernel(unsigned int* __restrict__ p, size_t n) {
  size_t i = (size_t)blockIdx.x * 256 + threadIdx.x;
  if (i < n) p[i] = 0u;
}

__global__ void f2bf_kernel(const float* __restrict__ in,
                            unsigned short* __restrict__ out, size_t n) {
  size_t i = (size_t)blockIdx.x * 256 + threadIdx.x;
  if (i < n) out[i] = f2bf_bits(in[i]);
}

__global__ void bias_add_kernel(const float* __restrict__ a,
                                const float* __restrict__ b,
                                float* __restrict__ o, int n) {
  int i = blockIdx.x * 256 + threadIdx.x;
  if (i < n) o[i] = a[i] + b[i];
}

// Swizzle fp32 weight W[N,K] (row-major, g = h . W^T) into bf16 WMMA
// B-fragment order: out[((nt*(K/32)+kc)*512) + lane*16 + e], where
//   col  j = nt*16 + (lane&15)
//   k      = kc*32 + (lane>=16 ? 16 : 0) + e
__global__ void swizzle_kernel(const float* __restrict__ W, int K,
                               unsigned short* __restrict__ out, size_t total) {
  size_t idx = (size_t)blockIdx.x * 256 + threadIdx.x;
  if (idx >= total) return;
  const int    e    = (int)(idx & 15);
  const int    lane = (int)((idx >> 4) & 31);
  const size_t blk  = idx >> 9;
  const int    kch  = K >> 5;
  const size_t nt   = blk / kch;
  const int    kc   = (int)(blk - nt * kch);
  const int    n    = (int)(nt << 4) + (lane & 15);
  const int    k    = (kc << 5) + ((lane >> 4) << 4) + e;
  out[idx] = f2bf_bits(W[(size_t)n * K + k]);
}

// ---- large input-projection GEMM: out[M,4096](bf16) = A[M,K](bf16) x W^T + bias
// WG = 256 thr (8 waves, 2x4): tile 128(M) x 128(N); wave = 4 Mtiles x 2 Ntiles.
__global__ __launch_bounds__(256) void gemm_xp_kernel(
    const bf16* __restrict__ A, int lda,
    const bf16* __restrict__ WF, int kchunks,
    const float* __restrict__ bias,
    unsigned short* __restrict__ out) {
  const int lane = threadIdx.x & 31;
  const int wid  = threadIdx.x >> 5;
  const int wm   = wid >> 2, wn = wid & 3;
  const int m0   = blockIdx.y * 128 + wm * 64;
  const int nt0  = blockIdx.x * 8 + wn * 2;

  v8f acc[4][2] = {};
  for (int kc = 0; kc < kchunks; ++kc) {
    v16bf a[4];
#pragma unroll
    for (int mt = 0; mt < 4; ++mt)
      a[mt] = load_A_frag(A, (size_t)lda, m0 + mt * 16, kc << 5, lane);
#pragma unroll
    for (int nn = 0; nn < 2; ++nn) {
      v16bf b = load_B_frag(WF, nt0 + nn, kc, kchunks, lane);
#pragma unroll
      for (int mt = 0; mt < 4; ++mt)
        acc[mt][nn] = WMMA_BF16(a[mt], b, acc[mt][nn]);
    }
  }
  // Epilogue: C layout — elem r of v8f is (M = m0 + (lane>=16)*8 + r, N = lane&15)
#pragma unroll
  for (int nn = 0; nn < 2; ++nn) {
    const int   n  = ((nt0 + nn) << 4) + (lane & 15);
    const float bv = bias[n];
#pragma unroll
    for (int mt = 0; mt < 4; ++mt) {
      const int rb = m0 + mt * 16 + ((lane >> 4) << 3);
#pragma unroll
      for (int r = 0; r < 8; ++r)
        out[(size_t)(rb + r) * NG + n] = f2bf_bits(acc[mt][nn][r] + bv);
    }
  }
}

// ---- fused LSTM cell step -------------------------------------------------
// g = A1 x WF1^T + A2 x WF2^T (+ bias | + xp[t]);  i,f,g,o -> c,h update.
// Grid 64 (jt = j-slice of H), block 256 (8 waves; wave == Mtile, M=128 once).
// B fragments for all 4 gates are double-buffered through LDS in groups of 4
// k-chunks (16KB per buffer): cooperative global_load_b128 -> ds_store_b128
// staging of group g+1 overlaps WMMA consumption of group g.
__global__ __launch_bounds__(256) void cell_kernel(
    const bf16* __restrict__ A1, int lda1, int K1, const bf16* __restrict__ WF1,
    const bf16* __restrict__ A2, int lda2, int K2, const bf16* __restrict__ WF2,
    const float* __restrict__ bias,                 // [4096] or nullptr
    const unsigned short* __restrict__ addm,        // bf16 xp row block or nullptr
    long long addm_stride,                          // per-batch-row stride (elems)
    float* __restrict__ cst,                        // [128,1024] in-place
    unsigned short* __restrict__ hout, int ldh) {   // bf16 h out
  __shared__ __align__(16) char smem[2][16384];     // 2 x (4 kchunks x 4 gates x 1KB)
  const int tid  = threadIdx.x;
  const int lane = tid & 31;
  const int wid  = tid >> 5;
  const int m0   = wid << 4;                        // wave owns rows [m0, m0+16)
  const int jt   = blockIdx.x;                      // 0..63

  v8f acc[4] = {};

  auto run_seg = [&](const bf16* __restrict__ A, size_t lda, int K,
                     const bf16* __restrict__ WF) {
    const int kch  = K >> 5;                        // 32-wide k chunks
    const int ngrp = kch >> 2;                      // groups of 4 chunks
    // cooperative stage of one group (16KB): 1024 x 16B units, 4 per thread
    auto stage = [&](int buf, int kcg) {
#pragma unroll
      for (int i = 0; i < 4; ++i) {
        const int u   = tid + (i << 8);             // 0..1023
        const int kcl = u >> 8;                     // kchunk within group
        const int rem = u & 255;
        const int g   = rem >> 6;                   // gate
        const int win = rem & 63;                   // 16B unit within 1KB block
        const bf16* src = WF + (((size_t)(g * 64 + jt) * kch) + (kcg << 2) + kcl) * 512
                          + (win << 3);
        *(bf16x8*)(&smem[buf][(((kcl << 2) + g) << 10) + (win << 4)]) =
            *(const bf16x8*)src;
      }
    };
    stage(0, 0);
    __syncthreads();
    for (int kcg = 0; kcg < ngrp; ++kcg) {
      if (kcg + 1 < ngrp) stage((kcg + 1) & 1, kcg + 1);
      const int buf = kcg & 1;
#pragma unroll
      for (int kcl = 0; kcl < 4; ++kcl) {
        v16bf a = load_A_frag(A, lda, m0, ((kcg << 2) + kcl) << 5, lane);
#pragma unroll
        for (int g = 0; g < 4; ++g) {
          v16bf b = *(const v16bf*)(&smem[buf][(((kcl << 2) + g) << 10) + (lane << 5)]);
          acc[g] = WMMA_BF16(a, b, acc[g]);
        }
      }
      __syncthreads();   // staging of next group done; current buffer reusable
    }
  };

  if (K1 > 0) run_seg(A1, (size_t)lda1, K1, WF1);
  run_seg(A2, (size_t)lda2, K2, WF2);

  // Epilogue: elem r of v8f is (M = m0 + (lane>=16)*8 + r, N = jt*16 + (lane&15))
  const int j     = (jt << 4) + (lane & 15);
  const int rbase = m0 + ((lane >> 4) << 3);
#pragma unroll
  for (int r = 0; r < 8; ++r) {
    const int bb = rbase + r;
    float gi = acc[0][r], gf = acc[1][r], gg = acc[2][r], go = acc[3][r];
    if (bias) {
      gi += bias[j]; gf += bias[HH + j]; gg += bias[2 * HH + j]; go += bias[3 * HH + j];
    }
    if (addm) {
      const unsigned short* p = addm + (size_t)bb * addm_stride;
      gi += bf2f(p[j]);          gf += bf2f(p[HH + j]);
      gg += bf2f(p[2 * HH + j]); go += bf2f(p[3 * HH + j]);
    }
    const float i_ = sigm(gi), f_ = sigm(gf), g_ = tanhf(gg), o_ = sigm(go);
    const size_t ci = (size_t)bb * HH + j;
    const float c  = f_ * cst[ci] + i_ * g_;
    cst[ci] = c;
    hout[(size_t)bb * ldh + j] = f2bf_bits(o_ * tanhf(c));
  }
}

// ---- output projection: y[128,256] = h1[128,1024] x W_lin^T + b_lin -------
// Block 256 (8 waves = 8 Mtiles), grid 8 (2 Ntiles each). Writes fp32 y into
// the d_out slot and bf16 y into ybuf (next decode step's input segment).
__global__ __launch_bounds__(256) void out_proj_kernel(
    const bf16* __restrict__ A,
    const bf16* __restrict__ WF,
    const float* __restrict__ blin,
    float* __restrict__ yout, long long ldy,
    unsigned short* __restrict__ ybf) {
  const int lane = threadIdx.x & 31;
  const int wid  = threadIdx.x >> 5;
  const int m0   = wid * 16;
  const int nt0  = blockIdx.x * 2;

  v8f acc[2] = {};
  for (int kc = 0; kc < HH / 32; ++kc) {
    v16bf a = load_A_frag(A, HH, m0, kc << 5, lane);
#pragma unroll
    for (int nn = 0; nn < 2; ++nn) {
      v16bf b = load_B_frag(WF, nt0 + nn, kc, HH / 32, lane);
      acc[nn] = WMMA_BF16(a, b, acc[nn]);
    }
  }
#pragma unroll
  for (int nn = 0; nn < 2; ++nn) {
    const int   n  = ((nt0 + nn) << 4) + (lane & 15);
    const float bv = blin[n];
    const int   rb = m0 + ((lane >> 4) << 3);
#pragma unroll
    for (int r = 0; r < 8; ++r) {
      const float y = acc[nn][r] + bv;
      yout[(size_t)(rb + r) * ldy + n] = y;
      ybf[(size_t)(rb + r) * DD + n]   = f2bf_bits(y);
    }
  }
}

// ---------------------------------------------------------------------------

extern "C" void kernel_launch(void* const* d_in, const int* in_sizes, int n_in,
                              void* d_out, int out_size, void* d_ws, size_t ws_size,
                              hipStream_t stream) {
  (void)in_sizes; (void)n_in; (void)out_size; (void)ws_size;
  const float* x     = (const float*)d_in[0];
  const float* W_ih0 = (const float*)d_in[1];
  const float* W_hh0 = (const float*)d_in[2];
  const float* b_ih0 = (const float*)d_in[3];
  const float* b_hh0 = (const float*)d_in[4];
  const float* W_ih1 = (const float*)d_in[5];
  const float* W_hh1 = (const float*)d_in[6];
  const float* b_ih1 = (const float*)d_in[7];
  const float* b_hh1 = (const float*)d_in[8];
  const float* W_lin = (const float*)d_in[9];
  const float* b_lin = (const float*)d_in[10];
  float* out = (float*)d_out;     // [B, 60, D] fp32

  // ---- workspace layout (~610 MB) ----
  char*  w   = (char*)d_ws;
  size_t off = 0;
  auto take = [&](size_t bytes) -> char* {
    char* p = w + off;
    off = (off + bytes + 255) & ~(size_t)255;
    return p;
  };
  unsigned short* xp0   = (unsigned short*)take((size_t)MTOT * NG * 2);  // 256 MB
  unsigned short* xp1   = (unsigned short*)take((size_t)MTOT * NG * 2);  // 256 MB
  unsigned short* out0  = (unsigned short*)take((size_t)MTOT * HH * 2);  //  64 MB
  unsigned short* xbf   = (unsigned short*)take((size_t)MTOT * DD * 2);  //  16 MB
  unsigned short* WFih0 = (unsigned short*)take((size_t)NG * DD * 2);
  unsigned short* WFhh0 = (unsigned short*)take((size_t)NG * HH * 2);
  unsigned short* WFih1 = (unsigned short*)take((size_t)NG * HH * 2);
  unsigned short* WFhh1 = (unsigned short*)take((size_t)NG * HH * 2);
  unsigned short* WFlin = (unsigned short*)take((size_t)DD * HH * 2);
  float*          bias0 = (float*)take((size_t)NG * 4);
  float*          bias1 = (float*)take((size_t)NG * 4);
  float*          c0    = (float*)take((size_t)BB * HH * 4);
  float*          c1    = (float*)take((size_t)BB * HH * 4);
  unsigned short* h0a   = (unsigned short*)take((size_t)BB * HH * 2);
  unsigned short* h0b   = (unsigned short*)take((size_t)BB * HH * 2);
  unsigned short* h1a   = (unsigned short*)take((size_t)BB * HH * 2);
  unsigned short* h1b   = (unsigned short*)take((size_t)BB * HH * 2);
  unsigned short* hz    = (unsigned short*)take((size_t)BB * HH * 2);   // all-zero h
  unsigned short* ybuf  = (unsigned short*)take((size_t)BB * DD * 2);

  auto zero_words = [&](void* p, size_t words) {
    zero_kernel<<<dim3((unsigned)((words + 255) / 256)), 256, 0, stream>>>(
        (unsigned int*)p, words);
  };
  zero_words(c0, (size_t)BB * HH);
  zero_words(c1, (size_t)BB * HH);
  zero_words(hz, (size_t)BB * HH / 2);

  // ---- one-time weight swizzle + bias fold + x->bf16 ----
  auto swz = [&](const float* Wsrc, int N, int K, unsigned short* dst) {
    size_t n = (size_t)N * K;
    swizzle_kernel<<<dim3((unsigned)((n + 255) / 256)), 256, 0, stream>>>(Wsrc, K, dst, n);
  };
  swz(W_ih0, NG, DD, WFih0);
  swz(W_hh0, NG, HH, WFhh0);
  swz(W_ih1, NG, HH, WFih1);
  swz(W_hh1, NG, HH, WFhh1);
  swz(W_lin, DD, HH, WFlin);
  bias_add_kernel<<<dim3(NG / 256), 256, 0, stream>>>(b_ih0, b_hh0, bias0, NG);
  bias_add_kernel<<<dim3(NG / 256), 256, 0, stream>>>(b_ih1, b_hh1, bias1, NG);
  f2bf_kernel<<<dim3((unsigned)(((size_t)MTOT * DD + 255) / 256)), 256, 0, stream>>>(
      x, xbf, (size_t)MTOT * DD);

  const dim3 cgrid(HH / 16), cblk(256);
  auto cell = [&](const unsigned short* A1, int lda1, int K1, const unsigned short* WF1,
                  const unsigned short* A2, int lda2, int K2, const unsigned short* WF2,
                  const float* bias, const unsigned short* addm, long long addm_stride,
                  float* cst, unsigned short* hout, int ldh) {
    cell_kernel<<<cgrid, cblk, 0, stream>>>(
        (const bf16*)A1, lda1, K1, (const bf16*)WF1,
        (const bf16*)A2, lda2, K2, (const bf16*)WF2,
        bias, addm, addm_stride, cst, hout, ldh);
  };

  // ---- layer 0: xp0 = x . W_ih0^T + (b_ih0 + b_hh0), then 256-step scan ----
  gemm_xp_kernel<<<dim3(NG / 128, MTOT / 128), 256, 0, stream>>>(
      (const bf16*)xbf, DD, (const bf16*)WFih0, DD / 32, bias0, xp0);
  for (int t = 0; t < TT; ++t) {
    const unsigned short* hprev = (t == 0) ? hz : (out0 + (size_t)(t - 1) * HH);
    const int ldap = (t == 0) ? HH : TT * HH;
    cell(nullptr, 0, 0, nullptr,
         hprev, ldap, HH, WFhh0,
         nullptr, xp0 + (size_t)t * NG, (long long)TT * NG,
         c0, out0 + (size_t)t * HH, TT * HH);
  }

  // ---- layer 1: xp1 = out0 . W_ih1^T + (b_ih1 + b_hh1), then scan ----
  gemm_xp_kernel<<<dim3(NG / 128, MTOT / 128), 256, 0, stream>>>(
      (const bf16*)out0, HH, (const bf16*)WFih1, HH / 32, bias1, xp1);
  for (int t = 0; t < TT; ++t) {
    const unsigned short* hin = (t == 0) ? hz : ((t & 1) ? h1a : h1b);
    unsigned short*       ho  = (t & 1) ? h1b : h1a;     // t=255 -> h1b (final)
    cell(nullptr, 0, 0, nullptr,
         hin, HH, HH, WFhh1,
         nullptr, xp1 + (size_t)t * NG, (long long)TT * NG,
         c1, ho, HH);
  }

  // ---- y_last = h1_T . W_lin^T + b_lin  (slot 0 of d_out) ----
  out_proj_kernel<<<dim3(DD / 32), 256, 0, stream>>>(
      (const bf16*)h1b, (const bf16*)WFlin, b_lin, out, 60LL * DD, ybuf);

  // ---- 59 autoregressive decode steps ----
  for (int s = 0; s < DEC; ++s) {
    const unsigned short* h0in; int ld0;
    if (s == 0) { h0in = out0 + (size_t)(TT - 1) * HH; ld0 = TT * HH; }
    else        { h0in = (s & 1) ? h0a : h0b;          ld0 = HH;      }
    unsigned short* h0out = (s & 1) ? h0b : h0a;
    cell(ybuf, DD, DD, WFih0,                     // [y | h0] x [W_ih0 | W_hh0]
         h0in, ld0, HH, WFhh0,
         bias0, nullptr, 0,
         c0, h0out, HH);

    const unsigned short* h1in  = (s & 1) ? h1a : h1b;
    unsigned short*       h1out = (s & 1) ? h1b : h1a;
    cell(h0out, HH, HH, WFih1,                    // [h0 | h1] x [W_ih1 | W_hh1]
         h1in, HH, HH, WFhh1,
         bias1, nullptr, 0,
         c1, h1out, HH);

    out_proj_kernel<<<dim3(DD / 32), 256, 0, stream>>>(
        (const bf16*)h1out, (const bf16*)WFlin, b_lin,
        out + (size_t)(s + 1) * DD, 60LL * DD, ybuf);
  }
}
